// MultiHeadCrossModalAttention_51264729645558
// MI455X (gfx1250) — matrix operand order repaired and checked by reading, not compiled, not run on hardware
//
#include <hip/hip_runtime.h>
#include <hip/hip_bf16.h>

typedef __attribute__((ext_vector_type(16))) _Float16 v16h;
typedef __attribute__((ext_vector_type(8)))  float    v8f;
typedef __attribute__((ext_vector_type(4)))  int      v4i;

#define AS1 __attribute__((address_space(1)))
#define AS3 __attribute__((address_space(3)))

#define B_     2
#define TQ_    1024
#define TK_    4096
#define DIM_   1024
#define NH_    16
#define HD_    64
#define SCALE_ 0.125f

// ---------------------------------------------------------------------------
// Async global->LDS staging (gfx1250 GLOBAL_LOAD_ASYNC_TO_LDS_B128, ASYNCcnt)
// Builtin signature (from clang diagnostic): (v4i AS1*, v4i AS3*, imm, imm)
// ---------------------------------------------------------------------------
#if __has_builtin(__builtin_amdgcn_global_load_async_to_lds_b128)
#define HAVE_ASYNC_LDS 1
__device__ inline void async_ld_b128(const _Float16* g, _Float16* l) {
    __builtin_amdgcn_global_load_async_to_lds_b128(
        (AS1 v4i*)g, (AS3 v4i*)l, 0, 0);
}
__device__ inline void wait_async0() {
#if __has_builtin(__builtin_amdgcn_s_wait_asynccnt)
    __builtin_amdgcn_s_wait_asynccnt(0);
#else
    asm volatile("s_wait_asynccnt 0" ::: "memory");
#endif
}
#else
#define HAVE_ASYNC_LDS 0
__device__ inline void wait_async0() {}
#endif

// stage 16 consecutive elements into LDS (f32 converts via VGPRs; f16 async)
__device__ inline void stage16(const float* src, _Float16* dst) {
    #pragma unroll
    for (int i = 0; i < 16; ++i) dst[i] = (_Float16)src[i];
}
__device__ inline void stage16(const _Float16* src, _Float16* dst) {
#if HAVE_ASYNC_LDS
    async_ld_b128(src, dst);            // 8 halves = 16B
    async_ld_b128(src + 8, dst + 8);
#else
    #pragma unroll
    for (int i = 0; i < 16; ++i) dst[i] = src[i];
#endif
}
// stage 8 consecutive elements
__device__ inline void stage8(const float* src, _Float16* dst) {
    #pragma unroll
    for (int i = 0; i < 8; ++i) dst[i] = (_Float16)src[i];
}
__device__ inline void stage8(const _Float16* src, _Float16* dst) {
#if HAVE_ASYNC_LDS
    async_ld_b128(src, dst);
#else
    #pragma unroll
    for (int i = 0; i < 8; ++i) dst[i] = src[i];
#endif
}

union FragU { v16h v; unsigned int u[8]; };

// A fragment (16x32 f16, wave32 ISA layout): lane row = l&15, kA = (l>>4)*8;
// VGPR j<4 -> K = kA+2j, VGPR j>=4 -> K = 16+kA+2(j-4). ar = &row[0].
__device__ inline void frag_a(const _Float16* ar, int kA, FragU& f) {
    f.u[0] = *(const unsigned int*)(ar + kA + 0);
    f.u[1] = *(const unsigned int*)(ar + kA + 2);
    f.u[2] = *(const unsigned int*)(ar + kA + 4);
    f.u[3] = *(const unsigned int*)(ar + kA + 6);
    f.u[4] = *(const unsigned int*)(ar + 16 + kA + 0);
    f.u[5] = *(const unsigned int*)(ar + 16 + kA + 2);
    f.u[6] = *(const unsigned int*)(ar + 16 + kA + 4);
    f.u[7] = *(const unsigned int*)(ar + 16 + kA + 6);
}

// B fragment (32x16 f16): lane col = l&15, kB = (l>>4)*8; VGPR j<4 -> K = kB+2j, j>=4 -> K = 16+kB+2(j-4).
__device__ inline void frag_b(const _Float16* br, int kB, FragU& f) {
    #pragma unroll
    for (int j = 0; j < 8; ++j)
        f.u[j] = *(const unsigned int*)(br + kB + 2 * (j & 3) + ((j >> 2) << 4));
}

__device__ inline v8f wmma_f16(const FragU& a, const FragU& b, v8f c) {
    return __builtin_amdgcn_wmma_f32_16x16x32_f16(
        false, a.v, false, b.v, (short)0, c, false, false);
}

// ---------------------------------------------------------------------------
// Generic tiled GEMM: out = alpha * (X @ B^T) + bias
//   X: [M,K] (ldx)  TX in {f32,f16};  B: [N,K] (ldb)  TB in {f32,f16}
//   out: [M,N] (ldo), or [N,M] if TRANS.
// Per-z offsets: off = (z/H)*s1 + (z%H)*s2  (batch+head slicing).
// WG: 256 threads = 8 waves. Macro tile 128(M) x 64(N), K chunk 32.
// Waves arranged 4(M) x 2(N); each wave computes a 2x2 grid of 16x16
// subtiles -> 4 x v_wmma_f32_16x16x32_f16 per k-chunk with fragment reuse.
// f16 operands are staged with async global->LDS b128 copies (ASYNCcnt).
// ---------------------------------------------------------------------------
template <typename TX, typename TB, typename TO, bool TRANS>
__global__ __launch_bounds__(256)
void gemm_xwT(const TX* __restrict__ X, int ldx, long long xs1, long long xs2,
              const TB* __restrict__ Bm, int ldb, long long bs1, long long bs2,
              TO* __restrict__ out, int ldo, long long os1, long long os2,
              const float* __restrict__ bias, float alpha, int K, int H)
{
    // stride 40 halves = 80B rows: every b128 LDS destination is 16B-aligned
    __shared__ __align__(16) _Float16 sA[128][40];
    __shared__ __align__(16) _Float16 sB[64][40];

    const int z  = blockIdx.z;
    const int zb = z / H;
    const int zh = z % H;
    const size_t xoff = (size_t)(zb * xs1 + (long long)zh * xs2);
    const size_t boff = (size_t)(zb * bs1 + (long long)zh * bs2);
    const size_t ooff = (size_t)(zb * os1 + (long long)zh * os2);

    const int m0 = blockIdx.x * 128;
    const int n0 = blockIdx.y * 64;

    const int tid   = threadIdx.x;
    const int lane  = tid & 31;
    const int wave  = tid >> 5;
    const int sub_m = (wave & 3) * 16;   // wave grid 4 (M) x 2 (N)
    const int sub_n = (wave >> 2) * 16;
    const int lrow  = lane & 15;
    const int hi    = lane >> 4;
    const int kA    = hi * 8;
    const int kB    = hi * 8;

    // cooperative-load indices
    const int a_row = tid >> 1;          // 0..127, 16 consecutive k each
    const int a_kst = (tid & 1) * 16;
    const int b_row = tid >> 2;          // 0..63, 8 consecutive k each
    const int b_kst = (tid & 3) * 8;

    constexpr bool anyAsync = HAVE_ASYNC_LDS && (sizeof(TX) == 2 || sizeof(TB) == 2);

    v8f acc00 = {}, acc01 = {}, acc10 = {}, acc11 = {};

    for (int kc = 0; kc < K; kc += 32) {
        // --- stage A tile: 128 rows x 32 k ---
        {
            const TX* src = X + xoff + (size_t)(m0 + a_row) * ldx + kc + a_kst;
            stage16(src, &sA[a_row][a_kst]);
            if (kc + 32 < K) __builtin_prefetch(src + 32, 0, 3);
        }
        // --- stage B tile: 64 n-rows x 32 k ---
        {
            const TB* src = Bm + boff + (size_t)(n0 + b_row) * ldb + kc + b_kst;
            stage8(src, &sB[b_row][b_kst]);
            if (kc + 32 < K) __builtin_prefetch(src + 32, 0, 3);
        }
        if (anyAsync) wait_async0();     // drain ASYNCcnt before LDS reads
        __syncthreads();

        FragU fa0, fa1, fb0, fb1;
        frag_a(&sA[sub_m +  0 + lrow][0], kA, fa0);
        frag_a(&sA[sub_m + 64 + lrow][0], kA, fa1);
        frag_b(&sB[sub_n +  0 + lrow][0], kB, fb0);
        frag_b(&sB[sub_n + 32 + lrow][0], kB, fb1);

        acc00 = wmma_f16(fa0, fb0, acc00);
        acc01 = wmma_f16(fa0, fb1, acc01);
        acc10 = wmma_f16(fa1, fb0, acc10);
        acc11 = wmma_f16(fa1, fb1, acc11);

        __syncthreads();
    }

    // --- write D: element (M,N) at lane = N + 16*(M>>3), vgpr = M&7 ---
    const int gn0 = n0 + sub_n + lrow;
    const int gn1 = gn0 + 32;
    const float bv0 = bias ? bias[gn0] : 0.0f;
    const float bv1 = bias ? bias[gn1] : 0.0f;
    #pragma unroll
    for (int r = 0; r < 8; ++r) {
        const int gmA = m0 + sub_m + hi * 8 + r;        // mi = 0
        const int gmB = gmA + 64;                       // mi = 1
        const float v00 = acc00[r] * alpha + bv0;
        const float v01 = acc01[r] * alpha + bv1;
        const float v10 = acc10[r] * alpha + bv0;
        const float v11 = acc11[r] * alpha + bv1;
        if (TRANS) {
            out[ooff + (size_t)gn0 * ldo + gmA] = (TO)v00;
            out[ooff + (size_t)gn1 * ldo + gmA] = (TO)v01;
            out[ooff + (size_t)gn0 * ldo + gmB] = (TO)v10;
            out[ooff + (size_t)gn1 * ldo + gmB] = (TO)v11;
        } else {
            out[ooff + (size_t)gmA * ldo + gn0] = (TO)v00;
            out[ooff + (size_t)gmA * ldo + gn1] = (TO)v01;
            out[ooff + (size_t)gmB * ldo + gn0] = (TO)v10;
            out[ooff + (size_t)gmB * ldo + gn1] = (TO)v11;
        }
    }
}

// ---------------------------------------------------------------------------
// Row softmax over 4096-wide rows, in place. One 256-thread WG per row.
// ---------------------------------------------------------------------------
__global__ __launch_bounds__(256) void softmax_rows(float* __restrict__ attn)
{
    __shared__ float red[256];
    const int tid = threadIdx.x;
    float* row = attn + (size_t)blockIdx.x * TK_;

    float v[16];
    float m = -3.402823466e38f;
    #pragma unroll
    for (int i = 0; i < 16; ++i) { v[i] = row[i * 256 + tid]; m = fmaxf(m, v[i]); }
    red[tid] = m; __syncthreads();
    for (int s = 128; s > 0; s >>= 1) {
        if (tid < s) red[tid] = fmaxf(red[tid], red[tid + s]);
        __syncthreads();
    }
    m = red[0]; __syncthreads();

    float sum = 0.0f;
    #pragma unroll
    for (int i = 0; i < 16; ++i) { v[i] = __expf(v[i] - m); sum += v[i]; }
    red[tid] = sum; __syncthreads();
    for (int s = 128; s > 0; s >>= 1) {
        if (tid < s) red[tid] += red[tid + s];
        __syncthreads();
    }
    const float inv = 1.0f / red[0];
    #pragma unroll
    for (int i = 0; i < 16; ++i) row[i * 256 + tid] = v[i] * inv;
}

// ---------------------------------------------------------------------------
// out = LayerNorm(query + proj) * gamma + beta. One WG per 1024-wide row.
// ---------------------------------------------------------------------------
__global__ __launch_bounds__(256)
void residual_ln(const float* __restrict__ q, const float* __restrict__ proj,
                 const float* __restrict__ gamma, const float* __restrict__ beta,
                 float* __restrict__ out)
{
    __shared__ float r1[256];
    __shared__ float r2[256];
    const int tid = threadIdx.x;
    const size_t base = (size_t)blockIdx.x * DIM_;

    float x[4];
    float s = 0.0f, ss = 0.0f;
    #pragma unroll
    for (int i = 0; i < 4; ++i) {
        const int j = i * 256 + tid;
        x[i] = q[base + j] + proj[base + j];
        s += x[i]; ss += x[i] * x[i];
    }
    r1[tid] = s; r2[tid] = ss; __syncthreads();
    for (int st = 128; st > 0; st >>= 1) {
        if (tid < st) { r1[tid] += r1[tid + st]; r2[tid] += r2[tid + st]; }
        __syncthreads();
    }
    const float mean = r1[0] * (1.0f / DIM_);
    const float var  = r2[0] * (1.0f / DIM_) - mean * mean;
    const float inv  = rsqrtf(var + 1e-5f);
    #pragma unroll
    for (int i = 0; i < 4; ++i) {
        const int j = i * 256 + tid;
        out[base + j] = (x[i] - mean) * inv * gamma[j] + beta[j];
    }
}

// ---------------------------------------------------------------------------
extern "C" void kernel_launch(void* const* d_in, const int* in_sizes, int n_in,
                              void* d_out, int out_size, void* d_ws, size_t ws_size,
                              hipStream_t stream)
{
    const float* query = (const float*)d_in[0];
    const float* key   = (const float*)d_in[1];
    const float* value = (const float*)d_in[2];
    const float* Wq    = (const float*)d_in[3];
    const float* bq    = (const float*)d_in[4];
    const float* Wk    = (const float*)d_in[5];
    const float* bk    = (const float*)d_in[6];
    const float* Wv    = (const float*)d_in[7];
    const float* bv    = (const float*)d_in[8];
    const float* Wo    = (const float*)d_in[9];
    const float* bo    = (const float*)d_in[10];
    const float* gamma = (const float*)d_in[11];
    const float* beta  = (const float*)d_in[12];

    float* out_final = (float*)d_out;                                   // [B,Tq,D]
    float* attn      = (float*)d_out + (size_t)B_ * TQ_ * DIM_;         // [B,H,Tq,Tk]

    // workspace layout (bytes)
    char* ws = (char*)d_ws;
    _Float16* Qp  = (_Float16*)(ws + 0);                        //  4 MB [B*Tq, D] f16
    _Float16* Kp  = (_Float16*)(ws + (4u  << 20));              // 16 MB [B*Tk, D] f16
    _Float16* Vt  = (_Float16*)(ws + (20u << 20));              // 16 MB [B, D, Tk] f16 (transposed)
    float*    ctx = (float*)   (ws + (36u << 20));              //  8 MB [B*Tq, D] f32
    float*    prj = (float*)   (ws + (44u << 20));              //  8 MB [B*Tq, D] f32

    const dim3 blk(256);
    const long long TqD  = (long long)TQ_ * DIM_;    // 1048576
    const long long TkD  = (long long)TK_ * DIM_;    // 4194304
    const long long TqTk = (long long)TQ_ * TK_;     // 4194304

    // 1) Q projection: Qp = query @ Wq^T + bq   (f32 in, f16 out)
    gemm_xwT<float, float, _Float16, false>
        <<<dim3((B_ * TQ_) / 128, DIM_ / 64, 1), blk, 0, stream>>>(
        query, DIM_, 0, 0, Wq, DIM_, 0, 0, Qp, DIM_, 0, 0, bq, 1.0f, DIM_, 1);

    // 2) K projection: Kp = key @ Wk^T + bk
    gemm_xwT<float, float, _Float16, false>
        <<<dim3((B_ * TK_) / 128, DIM_ / 64, 1), blk, 0, stream>>>(
        key, DIM_, 0, 0, Wk, DIM_, 0, 0, Kp, DIM_, 0, 0, bk, 1.0f, DIM_, 1);

    // 3) V projection, stored transposed per batch: Vt[b][d][t]
    gemm_xwT<float, float, _Float16, true>
        <<<dim3(TK_ / 128, DIM_ / 64, B_), blk, 0, stream>>>(
        value, DIM_, TkD, 0, Wv, DIM_, 0, 0, Vt, TK_, TkD, 0, bv, 1.0f, DIM_, 1);

    // 4) scores = SCALE * Qh @ Kh^T -> attn region (raw scores, f32);
    //    both operands f16 -> async LDS staging path
    gemm_xwT<_Float16, _Float16, float, false>
        <<<dim3(TQ_ / 128, TK_ / 64, B_ * NH_), blk, 0, stream>>>(
        Qp, DIM_, TqD, HD_,
        Kp, DIM_, TkD, HD_,
        attn, TK_, (long long)NH_ * TqTk, TqTk,
        nullptr, SCALE_, HD_, NH_);

    // 5) softmax over last dim, in place
    softmax_rows<<<dim3(B_ * NH_ * TQ_), blk, 0, stream>>>(attn);

    // 6) ctx[b, q, h*64+d] = attn @ V  (attn f32 X, Vt f16 B); N=64 in ONE
    //    tile -> the 537 MB attn tensor is read exactly once.
    gemm_xwT<float, _Float16, float, false>
        <<<dim3(TQ_ / 128, HD_ / 64, B_ * NH_), blk, 0, stream>>>(
        attn, TK_, (long long)NH_ * TqTk, TqTk,
        Vt, TK_, TkD, (long long)HD_ * TK_,
        ctx, DIM_, TqD, HD_,
        nullptr, 1.0f, TK_, NH_);

    // 7) output projection: prj = ctx @ Wo^T + bo
    gemm_xwT<float, float, float, false>
        <<<dim3((B_ * TQ_) / 128, DIM_ / 64, 1), blk, 0, stream>>>(
        ctx, DIM_, 0, 0, Wo, DIM_, 0, 0, prj, DIM_, 0, 0, bo, 1.0f, DIM_, 1);

    // 8) out = LayerNorm(query + prj)
    residual_ln<<<dim3(B_ * TQ_), blk, 0, stream>>>(query, prj, gamma, beta, out_final);
}
